// CustomLayer_22428319220577
// MI455X (gfx1250) — matrix operand order repaired and checked by reading, not compiled, not run on hardware
//
#include <hip/hip_runtime.h>
#include <hip/hip_bf16.h>
#include <stdint.h>

typedef __attribute__((ext_vector_type(16))) __bf16 v16bf;
typedef __attribute__((ext_vector_type(8)))  __bf16 bf16x8;
typedef __attribute__((ext_vector_type(2)))  __bf16 bf16x2;
typedef __attribute__((ext_vector_type(8)))  float  v8f;

#define GMIN (1.0f/40000.0f)
#define GMAX (1.0f/1000.0f)
#define RP   2.0f
#define EPSV 1e-8f
#define KDIM 1024
#define NDIM 1024

// ws layout (bytes)
#define WS_STAT_OFF 0            // float[8]: wmin, wmax, a, b, inv_a
#define WS_PMIN_OFF 64           // float[256]
#define WS_PMAX_OFF 1088         // float[256]
#define WS_B_OFF    4096         // 4 x 1M bf16 matrices (2MB each)
#define B_MAT_ELEMS (1024*1024)

__device__ __forceinline__ __bf16 f2bf(float f) {
    unsigned u = __builtin_bit_cast(unsigned, f);
    unsigned r = (u + 0x7FFFu + ((u >> 16) & 1u)) >> 16;
    unsigned short s = (unsigned short)r;
    return __builtin_bit_cast(__bf16, s);
}
__device__ __forceinline__ float bf2f(__bf16 h) {
    unsigned short s = __builtin_bit_cast(unsigned short, h);
    unsigned u = ((unsigned)s) << 16;
    return __builtin_bit_cast(float, u);
}

// ---------------- Kernel 1: per-block min/max of weight ----------------
__global__ void k_minmax(const float* __restrict__ w, char* __restrict__ ws) {
    __shared__ float smn[256], smx[256];
    const int t = threadIdx.x;
    const int g = blockIdx.x * 256 + t;
    float mn = 3.4e38f, mx = -3.4e38f;
    for (int i = g; i < KDIM * NDIM; i += 256 * 256) {
        float v = w[i];
        mn = fminf(mn, v); mx = fmaxf(mx, v);
    }
    smn[t] = mn; smx[t] = mx;
    __syncthreads();
    for (int s = 128; s > 0; s >>= 1) {
        if (t < s) { smn[t] = fminf(smn[t], smn[t + s]); smx[t] = fmaxf(smx[t], smx[t + s]); }
        __syncthreads();
    }
    if (t == 0) {
        ((float*)(ws + WS_PMIN_OFF))[blockIdx.x] = smn[0];
        ((float*)(ws + WS_PMAX_OFF))[blockIdx.x] = smx[0];
    }
}

// ---------------- Kernel 2: finalize stats, compute a / b / 1/a --------
__global__ void k_stats(char* __restrict__ ws) {
    __shared__ float smn[256], smx[256];
    const int t = threadIdx.x;
    smn[t] = ((const float*)(ws + WS_PMIN_OFF))[t];
    smx[t] = ((const float*)(ws + WS_PMAX_OFF))[t];
    __syncthreads();
    for (int s = 128; s > 0; s >>= 1) {
        if (t < s) { smn[t] = fminf(smn[t], smn[t + s]); smx[t] = fmaxf(smx[t], smx[t + s]); }
        __syncthreads();
    }
    if (t == 0) {
        float wmin = smn[0], wmax = smx[0];
        float a = (GMAX - GMIN) / (wmax - wmin);
        float* st = (float*)(ws + WS_STAT_OFF);
        st[0] = wmin; st[1] = wmax; st[2] = a;
        st[3] = GMIN - a * wmin;       // b
        st[4] = 1.0f / a;              // inv_a
    }
}

// ---------------- Kernel 3: build packed bf16 hi/lo B matrices ---------
// Fragment-major layout: elem = ((n_tile*32 + k_chunk)*32 + lane)*16 + (k%16)
// where lane = (n%16) + ((k>>4)&1)*16  (contiguous k = 16*half + k%16, same order as the A fragments).
__global__ void k_pack(const float* __restrict__ w, char* __restrict__ ws) {
    const float* st = (const float*)(ws + WS_STAT_OFF);
    const float wmin = st[0], wmax = st[1];
    const int idx = blockIdx.x * 256 + threadIdx.x;   // 0 .. 1M-1
    const int k = idx >> 10;
    const int n = idx & 1023;
    const float wv = w[idx];
    // affine map + 4-bit uniform quantization (t is shared between both)
    float tq = (wv - wmin) / (wmax - wmin);
    float G = rintf(tq * 15.0f) * (1.0f / 15.0f) * (GMAX - GMIN) + GMIN;
    // Jeong closed-form IR drop
    float rpar = RP * ((float)(KDIM - k) + (float)(n + 1));
    float Geff = 1.0f / (1.0f / G + rpar);
    // hi/lo bf16 split for split-precision GEMM
    __bf16 ghi = f2bf(G);
    __bf16 glo = f2bf(G - bf2f(ghi));
    __bf16 ehi = f2bf(Geff);
    __bf16 elo = f2bf(Geff - bf2f(ehi));

    const int n_tile = n >> 4, nin = n & 15;
    const int kc = k >> 5, halfk = (k >> 4) & 1, kin = k & 15;
    const size_t off = ((size_t)(n_tile * 32 + kc) * 32 + (nin + halfk * 16)) * 16 + kin;
    __bf16* b0 = (__bf16*)(ws + WS_B_OFF);
    b0[off]                   = ghi;
    b0[off + 1*B_MAT_ELEMS]   = glo;
    b0[off + 2*B_MAT_ELEMS]   = ehi;
    b0[off + 3*B_MAT_ELEMS]   = elo;
}

// ---------------- Kernel 4: fused split-bf16 WMMA GEMM + epilogue ------
// Block: 32 rows x 1024 cols. 16 waves; wave owns 64 cols (4 N-tiles) x 2 M-tiles.
// ~128 accumulator VGPRs/wave keeps total <256 (no VGPR-MSB addressing).
__global__ void __launch_bounds__(512, 1)
k_gemm(const float* __restrict__ x, const float* __restrict__ bias,
       const char* __restrict__ ws, float* __restrict__ out) {
    const float* st   = (const float*)(ws + WS_STAT_OFF);
    const __bf16* pgh = (const __bf16*)(ws + WS_B_OFF);
    const __bf16* pgl = pgh + 1 * B_MAT_ELEMS;
    const __bf16* peh = pgh + 2 * B_MAT_ELEMS;
    const __bf16* pel = pgh + 3 * B_MAT_ELEMS;

    __shared__ __bf16 shi[32 * 56];      // x hi tile, padded stride 56 (16B aligned, bank-spread)
    __shared__ __bf16 slo[32 * 56];      // x lo tile
    __shared__ float  srs[32 * 16];      // rowsum partials
    __shared__ float  sstat[32 * 16 * 6];// per-row per-wave {mnC,mxC,smC,mnI,mxI,smI}
    __shared__ float  sparam[32 * 2];    // per-row {alpha, beta}

    const int t    = threadIdx.x;
    const int wave = t >> 5;
    const int lane = t & 31;
    const int lh   = lane >> 4;     // lane half
    const int nin  = lane & 15;
    const int row0 = blockIdx.x * 32;

    // staging mapping: 512 threads = 32 rows x 16 threads, 2 floats each
    const int lrow = t >> 4;
    const int lcol = (t & 15) * 2;

    v8f accC[8] = {};   // Jeong currents, [mt*4 + t8]
    v8f accI[8] = {};   // ideal currents
    float rs = 0.0f;    // exact f32 row-sum partial

    #pragma unroll 1
    for (int kc = 0; kc < 32; ++kc) {
        // ---- stage x tile (f32 -> bf16 hi/lo) ----
        {
            const float2 v = *(const float2*)(x + (size_t)(row0 + lrow) * KDIM + kc * 32 + lcol);
            float a0 = v.x, a1 = v.y;
            __bf16 h0 = f2bf(a0), h1 = f2bf(a1);
            bf16x2 hv = {h0, h1};
            bf16x2 lv = {f2bf(a0 - bf2f(h0)), f2bf(a1 - bf2f(h1))};
            const int so = lrow * 56 + lcol;
            *(bf16x2*)(&shi[so]) = hv;
            *(bf16x2*)(&slo[so]) = lv;
            rs += a0 + a1;
        }
        __syncthreads();

        // ---- A fragments from LDS (contiguous k = 16*lh + s, same order as packed B) ----
        v16bf ahi[2], alo[2];
        #pragma unroll
        for (int mt = 0; mt < 2; ++mt) {
            const int base = (mt * 16 + nin) * 56 + lh * 16;
            bf16x8 p0 = *(const bf16x8*)(&shi[base]);
            bf16x8 p1 = *(const bf16x8*)(&shi[base + 8]);
            ahi[mt] = __builtin_shufflevector(p0, p1, 0,1,2,3,4,5,6,7,8,9,10,11,12,13,14,15);
            p0 = *(const bf16x8*)(&slo[base]);
            p1 = *(const bf16x8*)(&slo[base + 8]);
            alo[mt] = __builtin_shufflevector(p0, p1, 0,1,2,3,4,5,6,7,8,9,10,11,12,13,14,15);
        }
        __syncthreads();

        // ---- B fragments (coalesced b128 pairs) + 48 WMMAs ----
        #pragma unroll
        for (int t8 = 0; t8 < 4; ++t8) {
            const int n_tile = wave * 4 + t8;
            const size_t boff = ((size_t)(n_tile * 32 + kc) * 32 + lane) * 16;
            v16bf bgh = *(const v16bf*)(pgh + boff);
            v16bf bgl = *(const v16bf*)(pgl + boff);
            v16bf beh = *(const v16bf*)(peh + boff);
            v16bf bel = *(const v16bf*)(pel + boff);
            #pragma unroll
            for (int mt = 0; mt < 2; ++mt) {
                const int ai = mt * 4 + t8;
                accI[ai] = __builtin_amdgcn_wmma_f32_16x16x32_bf16(false, ahi[mt], false, bgh, (short)0, accI[ai], false, false);
                accI[ai] = __builtin_amdgcn_wmma_f32_16x16x32_bf16(false, alo[mt], false, bgh, (short)0, accI[ai], false, false);
                accI[ai] = __builtin_amdgcn_wmma_f32_16x16x32_bf16(false, ahi[mt], false, bgl, (short)0, accI[ai], false, false);
                accC[ai] = __builtin_amdgcn_wmma_f32_16x16x32_bf16(false, ahi[mt], false, beh, (short)0, accC[ai], false, false);
                accC[ai] = __builtin_amdgcn_wmma_f32_16x16x32_bf16(false, alo[mt], false, beh, (short)0, accC[ai], false, false);
                accC[ai] = __builtin_amdgcn_wmma_f32_16x16x32_bf16(false, ahi[mt], false, bel, (short)0, accC[ai], false, false);
            }
        }
    }

    // ---- per-row statistics (fused epilogue) ----
    srs[lrow * 16 + (t & 15)] = rs;

    #pragma unroll
    for (int mt = 0; mt < 2; ++mt) {
        #pragma unroll
        for (int v = 0; v < 8; ++v) {
            float mnC = 3.4e38f, mxC = -3.4e38f, smC = 0.0f;
            float mnI = 3.4e38f, mxI = -3.4e38f, smI = 0.0f;
            #pragma unroll
            for (int t8 = 0; t8 < 4; ++t8) {
                float c = accC[mt * 4 + t8][v];
                float i2 = accI[mt * 4 + t8][v];
                mnC = fminf(mnC, c); mxC = fmaxf(mxC, c); smC += c;
                mnI = fminf(mnI, i2); mxI = fmaxf(mxI, i2); smI += i2;
            }
            #pragma unroll
            for (int m = 1; m < 16; m <<= 1) {   // reduce across the 16-lane group
                mnC = fminf(mnC, __shfl_xor(mnC, m, 32));
                mxC = fmaxf(mxC, __shfl_xor(mxC, m, 32));
                smC += __shfl_xor(smC, m, 32);
                mnI = fminf(mnI, __shfl_xor(mnI, m, 32));
                mxI = fmaxf(mxI, __shfl_xor(mxI, m, 32));
                smI += __shfl_xor(smI, m, 32);
            }
            if (nin == 0) {
                const int r = mt * 16 + v + lh * 8;
                float* sp = &sstat[(r * 16 + wave) * 6];
                sp[0] = mnC; sp[1] = mxC; sp[2] = smC;
                sp[3] = mnI; sp[4] = mxI; sp[5] = smI;
            }
        }
    }
    __syncthreads();

    if (t < 32) {
        const int r = t;
        float rsum = 0.0f;
        #pragma unroll
        for (int j = 0; j < 16; ++j) rsum += srs[r * 16 + j];
        float mnC = 3.4e38f, mxC = -3.4e38f, smC = 0.0f;
        float mnI = 3.4e38f, mxI = -3.4e38f, smI = 0.0f;
        #pragma unroll
        for (int j = 0; j < 16; ++j) {
            const float* sp = &sstat[(r * 16 + j) * 6];
            mnC = fminf(mnC, sp[0]); mxC = fmaxf(mxC, sp[1]); smC += sp[2];
            mnI = fminf(mnI, sp[3]); mxI = fmaxf(mxI, sp[4]); smI += sp[5];
        }
        const float meanC = smC * (1.0f / 1024.0f);
        const float meanI = smI * (1.0f / 1024.0f);
        const float coeff = (mxI - mnI) / (mxC - mnC + EPSV);
        const float b = st[3], inv_a = st[4];
        // out = ((C - meanC)*coeff + meanI - rowsum*b)/a + bias = C*alpha + beta + bias
        sparam[r * 2 + 0] = coeff * inv_a;
        sparam[r * 2 + 1] = (meanI - meanC * coeff - rsum * b) * inv_a;
    }
    __syncthreads();

    // ---- correction + unmap + bias, store ----
    #pragma unroll
    for (int t8 = 0; t8 < 4; ++t8) {
        const int n = wave * 64 + t8 * 16 + nin;
        const float bv = bias[n];
        #pragma unroll
        for (int mt = 0; mt < 2; ++mt) {
            #pragma unroll
            for (int v = 0; v < 8; ++v) {
                const int r = mt * 16 + v + lh * 8;
                const float o = accC[mt * 4 + t8][v] * sparam[r * 2] + sparam[r * 2 + 1] + bv;
                out[(size_t)(row0 + r) * NDIM + n] = o;
            }
        }
    }
}

extern "C" void kernel_launch(void* const* d_in, const int* in_sizes, int n_in,
                              void* d_out, int out_size, void* d_ws, size_t ws_size,
                              hipStream_t stream) {
    (void)in_sizes; (void)n_in; (void)out_size; (void)ws_size;
    const float* x    = (const float*)d_in[0];
    const float* w    = (const float*)d_in[1];
    const float* bias = (const float*)d_in[2];
    float* out = (float*)d_out;
    char*  ws  = (char*)d_ws;

    k_minmax<<<256, 256, 0, stream>>>(w, ws);
    k_stats <<<1,   256, 0, stream>>>(ws);
    k_pack  <<<4096,256, 0, stream>>>(w, ws);
    k_gemm  <<<256, 512, 0, stream>>>(x, bias, ws, out);
}
